// Head_70935679861388
// MI455X (gfx1250) — matrix operand order repaired and checked by reading, not compiled, not run on hardware
//
#include <hip/hip_runtime.h>

typedef __attribute__((ext_vector_type(16))) __bf16 v16bf;
typedef __attribute__((ext_vector_type(8)))  __bf16 v8bf;
typedef __attribute__((ext_vector_type(8)))  float  v8f;

namespace {

constexpr int Tn = 256;    // sequence length
constexpr int Cn = 384;    // embedding dim
constexpr int Hn = 64;     // head size
constexpr int KP2 = 72;    // k-plane row stride (bf16): 144B, 16B-mult, 2-way-optimal b128 reads
constexpr int VTP = 264;   // vT-plane row stride (bf16): 528B, 16B-mult
constexpr int SP  = 40;    // per-wave staging row stride (bf16): 80B, 16B-mult
constexpr int WELEMS = 3 * Hn * Cn;   // elements per W plane in workspace

struct bf2 { __bf16 h; __bf16 l; };

__device__ __forceinline__ bf2 split(float f) {
  bf2 r;
  r.h = (__bf16)f;
  r.l = (__bf16)(f - (float)r.h);
  return r;
}

__device__ __forceinline__ v8f wmma_bf(v16bf a, v16bf b, v8f c) {
  return __builtin_amdgcn_wmma_f32_16x16x32_bf16(
      false, a, false, b, (short)0, c, false, false);
}

// fp32-emulating product: (ah+al) x (bh+bl) ~= ah*bh + ah*bl + al*bh
__device__ __forceinline__ v8f wmma3(v16bf ah, v16bf al, v16bf bh, v16bf bl, v8f c) {
  c = wmma_bf(ah, bh, c);
  c = wmma_bf(ah, bl, c);
  c = wmma_bf(al, bh, c);
  return c;
}

__device__ __forceinline__ float hmax16(float x) {
  x = fmaxf(x, __shfl_xor(x, 1, 32));
  x = fmaxf(x, __shfl_xor(x, 2, 32));
  x = fmaxf(x, __shfl_xor(x, 4, 32));
  x = fmaxf(x, __shfl_xor(x, 8, 32));
  return x;
}
__device__ __forceinline__ float hsum16(float x) {
  x += __shfl_xor(x, 1, 32);
  x += __shfl_xor(x, 2, 32);
  x += __shfl_xor(x, 4, 32);
  x += __shfl_xor(x, 8, 32);
  return x;
}

// Wave-internal LDS ordering (per-wave staging buffers, no workgroup barrier).
__device__ __forceinline__ void lds_fence_wave() {
  asm volatile("s_wait_dscnt 0" ::: "memory");
}

// A-fragment (16x32) hi/lo from f32 row (x in global): vector loads + split.
// Element e : col 16*hl+e.
__device__ __forceinline__ void make_afrag_x(const float* rowp, int hl,
                                             v16bf& ah, v16bf& al) {
  const float4* p0 = (const float4*)(rowp + 16 * hl);
  const float4* p1 = (const float4*)(rowp + 16 * hl + 8);
  float4 t0 = p0[0], t1 = p0[1], t2 = p1[0], t3 = p1[1];
  const float va[16] = {t0.x, t0.y, t0.z, t0.w, t1.x, t1.y, t1.z, t1.w,
                        t2.x, t2.y, t2.z, t2.w, t3.x, t3.y, t3.z, t3.w};
  #pragma unroll
  for (int e = 0; e < 16; ++e) {
    bf2 s = split(va[e]);
    ah[e] = s.h;
    al[e] = s.l;
  }
}

// A-fragment from a pre-split bf16 plane: two 16B runs, concatenated.
__device__ __forceinline__ v16bf load_afrag_bf(const __bf16* rowp, int hl) {
  v8bf a = *(const v8bf*)(rowp + 16 * hl);
  v8bf b = *(const v8bf*)(rowp + 16 * hl + 8);
  return __builtin_shufflevector(a, b, 0, 1, 2, 3, 4, 5, 6, 7,
                                 8, 9, 10, 11, 12, 13, 14, 15);
}

} // namespace

// ---- prep: split W into bf16 hi/lo planes of W^T ([3][H][C]) in workspace ----
__global__ __launch_bounds__(256) void split_w_kernel(
    const float* __restrict__ Wq, const float* __restrict__ Wk,
    const float* __restrict__ Wv, __bf16* __restrict__ hiT,
    __bf16* __restrict__ loT) {
  const int idx = blockIdx.x * 256 + threadIdx.x;
  if (idx >= WELEMS) return;
  const int m = idx / (Cn * Hn);
  const int c = (idx / Hn) % Cn;
  const int h = idx % Hn;
  const float* W = (m == 0) ? Wq : ((m == 1) ? Wk : Wv);
  const float f = W[(size_t)c * Hn + h];
  const __bf16 hi = (__bf16)f;
  const __bf16 lo = (__bf16)(f - (float)hi);
  const size_t o = ((size_t)m * Hn + h) * Cn + c;
  hiT[o] = hi;
  loT[o] = lo;
}

__global__ __launch_bounds__(256) void attn_head_kernel(
    const float* __restrict__ x, const __bf16* __restrict__ whiT,
    float* __restrict__ out) {
  // k planes: row-major [s][h]  -> score B-frags are contiguous 32B rows
  __shared__ __align__(16) __bf16 khi[Tn * KP2];       // 36,864 B
  __shared__ __align__(16) __bf16 klo[Tn * KP2];       // 36,864 B
  // v planes transposed [h][s]  -> output B-frags are contiguous 32B rows
  __shared__ __align__(16) __bf16 vThi[Hn * VTP];      // 33,792 B
  __shared__ __align__(16) __bf16 vTlo[Hn * VTP];      // 33,792 B
  // per-wave staging for q / P transposition (C/D layout -> A layout)
  __shared__ __align__(16) __bf16 shi[8 * 16 * SP];    // 10,240 B
  __shared__ __align__(16) __bf16 slo[8 * 16 * SP];    // 10,240 B

  const __bf16* wloT = whiT + WELEMS;

  const int b    = blockIdx.x;
  const int w    = threadIdx.x >> 5;
  const int lane = threadIdx.x & 31;
  const int hl   = lane >> 4;
  const int ln   = lane & 15;
  __bf16* swh = &shi[w * 16 * SP];
  __bf16* swl = &slo[w * 16 * SP];

  const int rts[2] = {w, 15 - w};   // balances causal work across waves

  const v8f vzero = {};
  v16bf qah[2][2], qal[2][2];       // q A-fragments, [tile][K-half]

  // ---------------- Stage A: q,k,v projections ----------------
  #pragma unroll
  for (int p = 0; p < 2; ++p) {
    const int rt = rts[p];
    const float* xr = x + ((size_t)b * Tn + rt * 16 + ln) * Cn;

    v8f acc[3][4];
    #pragma unroll
    for (int m = 0; m < 3; ++m)
      #pragma unroll
      for (int nt = 0; nt < 4; ++nt) acc[m][nt] = vzero;

    for (int kc = 0; kc < Cn / 32; ++kc) {
      const int kb = kc * 32;
      v16bf ah, al;
      make_afrag_x(xr + kb, hl, ah, al);

      #pragma unroll
      for (int m = 0; m < 3; ++m) {
        #pragma unroll
        for (int nt = 0; nt < 4; ++nt) {
          // B elem e: K = kb+16*hl+e, N = nt*16+ln; W^T makes this contiguous.
          const size_t o = ((size_t)(m * Hn + nt * 16 + ln)) * Cn + kb + 16 * hl;
          v16bf bh = *(const v16bf*)(whiT + o);
          v16bf bl = *(const v16bf*)(wloT + o);
          acc[m][nt] = wmma3(ah, al, bh, bl, acc[m][nt]);
        }
      }
    }

    // k, v -> pre-split bf16 LDS planes (v stored transposed)
    #pragma unroll
    for (int nt = 0; nt < 4; ++nt)
      #pragma unroll
      for (int j = 0; j < 8; ++j) {
        const int row = rt * 16 + j + 8 * hl;
        const int col = nt * 16 + ln;
        bf2 sk = split(acc[1][nt][j]);
        khi[row * KP2 + col] = sk.h;
        klo[row * KP2 + col] = sk.l;
        bf2 sv = split(acc[2][nt][j]);
        vThi[col * VTP + row] = sv.h;
        vTlo[col * VTP + row] = sv.l;
      }

    // q: C/D layout -> A layout via per-wave bf16 staging (two 32-col chunks)
    #pragma unroll
    for (int ch = 0; ch < 2; ++ch) {
      #pragma unroll
      for (int tt = 0; tt < 2; ++tt)
        #pragma unroll
        for (int j = 0; j < 8; ++j) {
          bf2 s = split(acc[0][ch * 2 + tt][j]);
          swh[(j + 8 * hl) * SP + tt * 16 + ln] = s.h;
          swl[(j + 8 * hl) * SP + tt * 16 + ln] = s.l;
        }
      lds_fence_wave();
      qah[p][ch] = load_afrag_bf(swh + ln * SP, hl);
      qal[p][ch] = load_afrag_bf(swl + ln * SP, hl);
      lds_fence_wave();
    }
  }

  __syncthreads();   // all k,v visible

  // ------------- Stage B: causal attention, online softmax -------------
  #pragma unroll
  for (int p = 0; p < 2; ++p) {
    const int rt = rts[p];
    v8f oacc[4];
    #pragma unroll
    for (int nt = 0; nt < 4; ++nt) oacc[nt] = vzero;
    float mrow[8], ssum[8];
    #pragma unroll
    for (int j = 0; j < 8; ++j) { mrow[j] = -1e30f; ssum[j] = 0.0f; }

    const int tsmax = rt >> 1;           // skip fully-masked tile pairs
    for (int ts = 0; ts <= tsmax; ++ts) {
      // score tiles t0 = 2ts, t1 = 2ts+1 : S = q @ k^T
      v8f s0 = {}, s1 = {};
      #pragma unroll
      for (int half = 0; half < 2; ++half) {
        const int ko = (ts * 32 + ln) * KP2 + half * 32 + 16 * hl;
        v16bf b0h = *(const v16bf*)&khi[ko];
        v16bf b0l = *(const v16bf*)&klo[ko];
        v16bf b1h = *(const v16bf*)&khi[ko + 16 * KP2];
        v16bf b1l = *(const v16bf*)&klo[ko + 16 * KP2];
        s0 = wmma3(qah[p][half], qal[p][half], b0h, b0l, s0);
        s1 = wmma3(qah[p][half], qal[p][half], b1h, b1l, s1);
      }

      // scale by H**0.5 = 8 (reproduces reference), mask, online softmax
      const int col0 = ts * 32 + ln;
      const int col1 = col0 + 16;
      float pr0[8], pr1[8], corr[8];
      #pragma unroll
      for (int j = 0; j < 8; ++j) {
        const int row = rt * 16 + j + 8 * hl;
        const float v0 = (col0 <= row) ? s0[j] * 8.0f : -1e30f;
        const float v1 = (col1 <= row) ? s1[j] * 8.0f : -1e30f;
        const float tmx = hmax16(fmaxf(v0, v1));
        const float mn  = fmaxf(mrow[j], tmx);
        const float c   = __expf(mrow[j] - mn);
        const float p0  = __expf(v0 - mn);
        const float p1  = __expf(v1 - mn);
        ssum[j] = ssum[j] * c + hsum16(p0 + p1);
        mrow[j] = mn;
        corr[j] = c;
        pr0[j]  = p0;
        pr1[j]  = p1;
      }
      #pragma unroll
      for (int nt = 0; nt < 4; ++nt)
        #pragma unroll
        for (int j = 0; j < 8; ++j)
          oacc[nt][j] = oacc[nt][j] * corr[j];

      // stage P (16x32, hi/lo bf16) per-wave, then O += P @ V
      #pragma unroll
      for (int j = 0; j < 8; ++j) {
        const int r = (j + 8 * hl) * SP;
        bf2 a = split(pr0[j]);
        swh[r + ln] = a.h;
        swl[r + ln] = a.l;
        bf2 c = split(pr1[j]);
        swh[r + 16 + ln] = c.h;
        swl[r + 16 + ln] = c.l;
      }
      lds_fence_wave();
      v16bf pah = load_afrag_bf(swh + ln * SP, hl);
      v16bf pal = load_afrag_bf(swl + ln * SP, hl);
      lds_fence_wave();

      #pragma unroll
      for (int nt = 0; nt < 4; ++nt) {
        // B elem e: K = ts*32+16*hl+e, N = nt*16+ln; vT makes this contiguous.
        const int vo = (nt * 16 + ln) * VTP + ts * 32 + 16 * hl;
        v16bf bh = *(const v16bf*)&vThi[vo];
        v16bf bl = *(const v16bf*)&vTlo[vo];
        oacc[nt] = wmma3(pah, pal, bh, bl, oacc[nt]);
      }
    }

    // normalize and store [B,T,H]
    float inv[8];
    #pragma unroll
    for (int j = 0; j < 8; ++j) inv[j] = 1.0f / ssum[j];
    #pragma unroll
    for (int nt = 0; nt < 4; ++nt)
      #pragma unroll
      for (int j = 0; j < 8; ++j) {
        const int row = rt * 16 + j + 8 * hl;
        out[((size_t)b * Tn + row) * Hn + nt * 16 + ln] = oacc[nt][j] * inv[j];
      }
  }
}

extern "C" void kernel_launch(void* const* d_in, const int* in_sizes, int n_in,
                              void* d_out, int out_size, void* d_ws, size_t ws_size,
                              hipStream_t stream) {
  const float* x  = (const float*)d_in[0];
  const float* Wq = (const float*)d_in[1];
  const float* Wk = (const float*)d_in[2];
  const float* Wv = (const float*)d_in[3];
  float* out = (float*)d_out;
  __bf16* wplanes = (__bf16*)d_ws;              // [hi | lo], 2*73,728 bf16 = 294,912 B

  const int B = in_sizes[0] / (Tn * Cn);        // 512

  split_w_kernel<<<dim3((WELEMS + 255) / 256), dim3(256), 0, stream>>>(
      Wq, Wk, Wv, wplanes, wplanes + WELEMS);
  attn_head_kernel<<<dim3(B), dim3(256), 0, stream>>>(x, wplanes, out);

  (void)n_in; (void)out_size; (void)ws_size;
}
